// TransolverBlock_58858231824692
// MI455X (gfx1250) — matrix operand order repaired and checked by reading, not compiled, not run on hardware
//
#include <hip/hip_runtime.h>
#include <hip/hip_bf16.h>

#define BB 4
#define NN 16384
#define CC 256
#define HHD 8
#define GG 64
#define DDIM 32
#define FFH 1024

typedef __attribute__((ext_vector_type(16))) __bf16 v16bf;
typedef __attribute__((ext_vector_type(8)))  float  v8f;
typedef __attribute__((ext_vector_type(4)))  unsigned int u32x4;

union Frag16 {
    v16bf v;
    unsigned short s[16];
    u32x4 q[2];
};

__device__ __forceinline__ unsigned short f2bf(float f) {
    union { float f; unsigned int u; } c; c.f = f;
    unsigned int u = c.u;
    return (unsigned short)((u + 0x7FFFu + ((u >> 16) & 1u)) >> 16);
}
__device__ __forceinline__ float bf2f(unsigned short s) {
    union { float f; unsigned int u; } c; c.u = ((unsigned int)s) << 16; return c.f;
}

// A-matrix 16x32 bf16 fragment: lane<16 holds row m, K = {0..15};
// lane>=16 holds row m, K = {16..31}.
__device__ __forceinline__ Frag16 load_a_frag(const unsigned short* base, int pitch, int m0, int k0) {
    const int lane = threadIdx.x & 31;
    const int m = m0 + (lane & 15);
    const int kh = (lane >> 4) << 4;
    const unsigned short* p = base + (size_t)m * pitch + k0 + kh;
    Frag16 f;
    f.q[0] = *(const u32x4*)(p);
    f.q[1] = *(const u32x4*)(p + 8);
    return f;
}

// B-matrix 32x16 bf16 fragment (pre-packed): lane = column, 16 contiguous K per lane.
__device__ __forceinline__ Frag16 load_b_frag(const unsigned short* packed, int tileIdx) {
    const int lane = threadIdx.x & 31;
    const unsigned short* p = packed + ((size_t)tileIdx << 9) + (lane << 4);
    Frag16 f;
    f.q[0] = *(const u32x4*)(p);
    f.q[1] = *(const u32x4*)(p + 8);
    return f;
}

__device__ __forceinline__ v8f wmma_bf16(const Frag16& a, const Frag16& b, v8f c) {
    return __builtin_amdgcn_wmma_f32_16x16x32_bf16(false, a.v, false, b.v, (short)0, c, false, false);
}

__device__ __forceinline__ v8f tile_mm(const unsigned short* aBase, int aPitch, int m0,
                                       const unsigned short* bPacked, int nTilesB, int nt,
                                       int kTiles) {
    v8f acc = {};
    for (int kt = 0; kt < kTiles; ++kt) {
        if (kt + 1 < kTiles)
            __builtin_prefetch(bPacked + (size_t)((kt + 1) * nTilesB + nt) * 512, 0, 1);
        Frag16 a = load_a_frag(aBase, aPitch, m0, kt * 32);
        Frag16 b = load_b_frag(bPacked, kt * nTilesB + nt);
        acc = wmma_bf16(a, b, acc);
    }
    return acc;
}

// ---------------- weight repack: f32 row-major (K x N) -> packed bf16 B-fragments ----------
__global__ void k_repack(const float* __restrict__ W, unsigned short* __restrict__ out,
                         int K, int N) {
    int idx = blockIdx.x * blockDim.x + threadIdx.x;
    int total = K * N;
    if (idx >= total) return;
    int i    = idx & 15;
    int lane = (idx >> 4) & 31;
    int tile = idx >> 9;
    int nT   = N >> 4;
    int kt   = tile / nT;
    int nt   = tile - kt * nT;
    int n = nt * 16 + (lane & 15);
    int k = kt * 32 + ((lane >> 4) << 4) + i;
    out[idx] = f2bf(W[(size_t)k * N + n]);
}

// ---------------- kernel 1: LN1, xm/fx projections, slice softmax, slice_token partials ----
__global__ void k_attn_pre(const float* __restrict__ x,
                           const float* __restrict__ ln1_s, const float* __restrict__ ln1_b,
                           const unsigned short* __restrict__ WxP,  const float* __restrict__ bx,
                           const unsigned short* __restrict__ WfxP, const float* __restrict__ bfx,
                           const unsigned short* __restrict__ WslP, const float* __restrict__ bslice,
                           const float* __restrict__ temp,
                           unsigned short* __restrict__ swG,
                           float* __restrict__ stnum, float* __restrict__ stden) {
    extern __shared__ char smem[];
    unsigned short* hL   = (unsigned short*)smem;          // 32 x 264 bf16
    unsigned short* fxm  = hL + 32 * 264;                  // 32 x 264 bf16
    unsigned short* swl  = fxm + 32 * 264;                 // 8 x 32 x 64 bf16
    float*          lgt  = (float*)(swl + 8 * 32 * 64);    // 32 x 64 f32

    const int b    = blockIdx.y;
    const int n0   = blockIdx.x * 32;
    const int tid  = threadIdx.x;
    const int wave = tid >> 5;
    const int lane = tid & 31;

    // LN1: one wave per token row
    for (int t = wave; t < 32; t += 8) {
        const float* xr = x + ((size_t)(b * NN + n0 + t)) * CC + lane * 8;
        float v[8]; float s = 0.f, ss = 0.f;
        for (int j = 0; j < 8; ++j) { v[j] = xr[j]; s += v[j]; ss += v[j] * v[j]; }
        for (int off = 16; off; off >>= 1) { s += __shfl_xor(s, off); ss += __shfl_xor(ss, off); }
        float mu  = s * (1.f / 256.f);
        float var = ss * (1.f / 256.f) - mu * mu;
        float rs  = rsqrtf(var + 1e-6f);
        for (int j = 0; j < 8; ++j) {
            int col = lane * 8 + j;
            hL[t * 264 + col] = f2bf((v[j] - mu) * rs * ln1_s[col] + ln1_b[col]);
        }
    }
    __syncthreads();

    // xm = h @ Wx + bx  (32x256 @ 256x256), WMMA, result -> fxm (bf16)
    for (int t = wave; t < 32; t += 8) {
        int mt = t >> 4, nt = t & 15;
        v8f acc = tile_mm(hL, 264, mt * 16, WxP, 16, nt, 8);
        int n  = nt * 16 + (lane & 15);
        int mo = (lane >> 4) << 3;
        float bias = bx[n];
        for (int r = 0; r < 8; ++r)
            fxm[(mt * 16 + mo + r) * 264 + n] = f2bf(acc[r] + bias);
    }
    __syncthreads();

    // per-head slice logits (WMMA) + softmax over G
    for (int hh = 0; hh < HHD; ++hh) {
        {
            int mt = wave >> 2, nt = wave & 3;      // 8 tiles = 8 waves
            v8f acc = {};
            Frag16 a = load_a_frag(fxm, 264, mt * 16, hh * 32);
            Frag16 bq = load_b_frag(WslP, nt);
            acc = wmma_bf16(a, bq, acc);
            float tc = temp[hh];
            tc = fminf(fmaxf(tc, 0.1f), 5.0f);
            float invt = 1.0f / tc;
            int n  = nt * 16 + (lane & 15);
            int mo = (lane >> 4) << 3;
            float bias = bslice[n];
            for (int r = 0; r < 8; ++r)
                lgt[(mt * 16 + mo + r) * 64 + n] = (acc[r] + bias) * invt;
        }
        __syncthreads();
        if (tid < 32) {
            int row = tid;
            float mx = -3e38f;
            for (int g = 0; g < GG; ++g) mx = fmaxf(mx, lgt[row * 64 + g]);
            float sum = 0.f;
            for (int g = 0; g < GG; ++g) sum += __expf(lgt[row * 64 + g] - mx);
            float inv = 1.f / sum;
            unsigned short* dst = swG + (((size_t)(b * HHD + hh) * NN) + n0 + row) * GG;
            for (int g = 0; g < GG; ++g) {
                unsigned short bb = f2bf(__expf(lgt[row * 64 + g] - mx) * inv);
                swl[(hh * 32 + row) * 64 + g] = bb;
                dst[g] = bb;
            }
        }
        __syncthreads();
    }

    // fx = h @ Wfx + bfx  -> overwrite fxm
    for (int t = wave; t < 32; t += 8) {
        int mt = t >> 4, nt = t & 15;
        v8f acc = tile_mm(hL, 264, mt * 16, WfxP, 16, nt, 8);
        int n  = nt * 16 + (lane & 15);
        int mo = (lane >> 4) << 3;
        float bias = bfx[n];
        for (int r = 0; r < 8; ++r)
            fxm[(mt * 16 + mo + r) * 264 + n] = f2bf(acc[r] + bias);
    }
    __syncthreads();

    // slice_token partial sums: st[h][g][d] += sum_n sw[h][n][g] * fx[n][h*32+d]
    for (int idx = tid; idx < HHD * GG * DDIM; idx += 256) {
        int hh = idx >> 11;
        int rem = idx & 2047;
        int g = rem >> 5, d = rem & 31;
        float acc = 0.f;
        for (int n2 = 0; n2 < 32; ++n2)
            acc += bf2f(swl[(hh * 32 + n2) * 64 + g]) * bf2f(fxm[n2 * 264 + hh * 32 + d]);
        atomicAdd(&stnum[((size_t)b * HHD + hh) * (GG * DDIM) + g * DDIM + d], acc);
    }
    for (int idx = tid; idx < HHD * GG; idx += 256) {
        int hh = idx >> 6, g = idx & 63;
        float acc = 0.f;
        for (int n2 = 0; n2 < 32; ++n2) acc += bf2f(swl[(hh * 32 + n2) * 64 + g]);
        atomicAdd(&stden[((size_t)b * HHD + hh) * GG + g], acc);
    }
}

// ---------------- kernel 2: per-(b,h) slice attention (tiny: 64x32) ------------------------
__global__ void k_attn_mid(const float* __restrict__ stnum, const float* __restrict__ stden,
                           const float* __restrict__ Wq, const float* __restrict__ Wk,
                           const float* __restrict__ Wv,
                           unsigned short* __restrict__ osP) {
    extern __shared__ char smem[];
    float* stL = (float*)smem;        // 64x32
    float* qL  = stL + 2048;          // 64x32
    float* kL  = qL + 2048;           // 64x32
    float* vL  = kL + 2048;           // 64x32
    float* aL  = vL + 2048;           // 64x64
    float* oL  = aL + 4096;           // 64x32

    const int bh  = blockIdx.x;
    const int tid = threadIdx.x;

    for (int idx = tid; idx < GG * DDIM; idx += 64) {
        int g = idx >> 5;
        stL[idx] = stnum[(size_t)bh * (GG * DDIM) + idx] / (stden[(size_t)bh * GG + g] + 1e-5f);
    }
    __syncthreads();

    for (int idx = tid; idx < GG * DDIM; idx += 64) {
        int g = idx >> 5, d = idx & 31;
        float aq = 0.f, ak = 0.f, av = 0.f;
        for (int e = 0; e < DDIM; ++e) {
            float s = stL[g * DDIM + e];
            aq += s * Wq[e * DDIM + d];
            ak += s * Wk[e * DDIM + d];
            av += s * Wv[e * DDIM + d];
        }
        qL[idx] = aq; kL[idx] = ak; vL[idx] = av;
    }
    __syncthreads();

    {   // attention row per thread (64 threads, 64 rows)
        int g = tid;
        const float scale = 0.17677669529663687f;  // 32^-0.5
        float mx = -3e38f;
        for (int j = 0; j < GG; ++j) {
            float s = 0.f;
            for (int d = 0; d < DDIM; ++d) s += qL[g * DDIM + d] * kL[j * DDIM + d];
            s *= scale;
            aL[g * GG + j] = s;
            mx = fmaxf(mx, s);
        }
        float sum = 0.f;
        for (int j = 0; j < GG; ++j) { float e = __expf(aL[g * GG + j] - mx); aL[g * GG + j] = e; sum += e; }
        float inv = 1.f / sum;
        for (int j = 0; j < GG; ++j) aL[g * GG + j] *= inv;
    }
    __syncthreads();

    for (int idx = tid; idx < GG * DDIM; idx += 64) {
        int g = idx >> 5, d = idx & 31;
        float acc = 0.f;
        for (int j = 0; j < GG; ++j) acc += aL[g * GG + j] * vL[j * DDIM + d];
        oL[idx] = acc;
    }
    __syncthreads();

    // emit out_slice (64x32: K=g, N=d) in packed B-fragment bf16 layout (kT=2, nT=2)
    for (int idx = tid; idx < 2048; idx += 64) {
        int i    = idx & 15;
        int lane = (idx >> 4) & 31;
        int tile = idx >> 9;
        int kt = tile >> 1, nt = tile & 1;
        int d = nt * 16 + (lane & 15);
        int g = kt * 32 + ((lane >> 4) << 4) + i;
        osP[(size_t)bh * 2048 + idx] = f2bf(oL[g * DDIM + d]);
    }
}

// ---------------- kernel 3: deslice (sw @ out_slice), Wo projection, residual --------------
__global__ void k_attn_post(const unsigned short* __restrict__ swG,
                            const unsigned short* __restrict__ osP,
                            const unsigned short* __restrict__ WoP, const float* __restrict__ bo,
                            const float* __restrict__ x, float* __restrict__ xout) {
    extern __shared__ char smem[];
    unsigned short* ox = (unsigned short*)smem;   // 32 x 264 bf16

    const int b    = blockIdx.y;
    const int n0   = blockIdx.x * 32;
    const int tid  = threadIdx.x;
    const int wave = tid >> 5;
    const int lane = tid & 31;

    // out_x[n, h*32+d] = sum_g sw[b,h,n,g] * out_slice[b,h,g,d]   (A from global, B packed)
    for (int t = wave; t < 32; t += 8) {          // 8 heads x (2 mt x 2 nt)
        int hh = t >> 2, mt = (t >> 1) & 1, nt = t & 1;
        v8f acc = {};
        for (int kt = 0; kt < 2; ++kt) {
            Frag16 a;
            {
                int m  = lane & 15;
                int kh = (lane >> 4) << 4;
                const unsigned short* p = swG + (((size_t)(b * HHD + hh) * NN) + n0 + mt * 16 + m) * GG
                                              + kt * 32 + kh;
                a.q[0] = *(const u32x4*)(p);
                a.q[1] = *(const u32x4*)(p + 8);
            }
            Frag16 bq = load_b_frag(osP + (size_t)(b * HHD + hh) * 2048, kt * 2 + nt);
            acc = wmma_bf16(a, bq, acc);
        }
        int n  = nt * 16 + (lane & 15);
        int mo = (lane >> 4) << 3;
        for (int r = 0; r < 8; ++r)
            ox[(mt * 16 + mo + r) * 264 + hh * 32 + n] = f2bf(acc[r]);
    }
    __syncthreads();

    // out = out_x @ Wo + bo + x
    for (int t = wave; t < 32; t += 8) {
        int mt = t >> 4, nt = t & 15;
        v8f acc = tile_mm(ox, 264, mt * 16, WoP, 16, nt, 8);
        int n  = nt * 16 + (lane & 15);
        int mo = (lane >> 4) << 3;
        float bias = bo[n];
        for (int r = 0; r < 8; ++r) {
            int row = mt * 16 + mo + r;
            size_t gi = ((size_t)(b * NN) + n0 + row) * CC + n;
            xout[gi] = acc[r] + bias + x[gi];
        }
    }
}

// ---------------- kernel 4: LN2 + FFN (gelu) + residual, in-place on d_out -----------------
__global__ void k_ffn(const float* __restrict__ ln2_s, const float* __restrict__ ln2_b,
                      const unsigned short* __restrict__ W1P, const float* __restrict__ b1,
                      const unsigned short* __restrict__ W2P, const float* __restrict__ b2,
                      float* __restrict__ xio) {
    extern __shared__ char smem[];
    unsigned short* h2 = (unsigned short*)smem;   // 32 x 264 bf16
    unsigned short* ff = h2 + 32 * 264;           // 32 x 1032 bf16

    const int b    = blockIdx.y;
    const int n0   = blockIdx.x * 32;
    const int tid  = threadIdx.x;
    const int wave = tid >> 5;
    const int lane = tid & 31;

    for (int t = wave; t < 32; t += 8) {
        const float* xr = xio + ((size_t)(b * NN + n0 + t)) * CC + lane * 8;
        float v[8]; float s = 0.f, ss = 0.f;
        for (int j = 0; j < 8; ++j) { v[j] = xr[j]; s += v[j]; ss += v[j] * v[j]; }
        for (int off = 16; off; off >>= 1) { s += __shfl_xor(s, off); ss += __shfl_xor(ss, off); }
        float mu  = s * (1.f / 256.f);
        float var = ss * (1.f / 256.f) - mu * mu;
        float rs  = rsqrtf(var + 1e-6f);
        for (int j = 0; j < 8; ++j) {
            int col = lane * 8 + j;
            h2[t * 264 + col] = f2bf((v[j] - mu) * rs * ln2_s[col] + ln2_b[col]);
        }
    }
    __syncthreads();

    // ff = gelu(h2 @ W1 + b1)   (32x256 @ 256x1024)
    for (int t = wave; t < 128; t += 8) {
        int mt = t >> 6, nt = t & 63;
        v8f acc = tile_mm(h2, 264, mt * 16, W1P, 64, nt, 8);
        int n  = nt * 16 + (lane & 15);
        int mo = (lane >> 4) << 3;
        float bias = b1[n];
        for (int r = 0; r < 8; ++r) {
            float u = acc[r] + bias;
            float g = 0.5f * u * (1.f + tanhf(0.7978845608028654f * (u + 0.044715f * u * u * u)));
            ff[(mt * 16 + mo + r) * 1032 + n] = f2bf(g);
        }
    }
    __syncthreads();

    // out = ff @ W2 + b2 + x_attn   (32x1024 @ 1024x256)
    for (int t = wave; t < 32; t += 8) {
        int mt = t >> 4, nt = t & 15;
        v8f acc = tile_mm(ff, 1032, mt * 16, W2P, 16, nt, 32);
        int n  = nt * 16 + (lane & 15);
        int mo = (lane >> 4) << 3;
        float bias = b2[n];
        for (int r = 0; r < 8; ++r) {
            int row = mt * 16 + mo + r;
            size_t gi = ((size_t)(b * NN) + n0 + row) * CC + n;
            xio[gi] = acc[r] + bias + xio[gi];
        }
    }
}

// -------------------------------------------------------------------------------------------
extern "C" void kernel_launch(void* const* d_in, const int* in_sizes, int n_in,
                              void* d_out, int out_size, void* d_ws, size_t ws_size,
                              hipStream_t stream) {
    const float* x      = (const float*)d_in[0];
    const float* ln1_s  = (const float*)d_in[1];
    const float* ln1_b  = (const float*)d_in[2];
    const float* Wfx    = (const float*)d_in[3];
    const float* bfx    = (const float*)d_in[4];
    const float* Wx     = (const float*)d_in[5];
    const float* bx     = (const float*)d_in[6];
    const float* Wslice = (const float*)d_in[7];
    const float* bslice = (const float*)d_in[8];
    const float* temp   = (const float*)d_in[9];
    const float* Wq     = (const float*)d_in[10];
    const float* Wk     = (const float*)d_in[11];
    const float* Wv     = (const float*)d_in[12];
    const float* Wo     = (const float*)d_in[13];
    const float* bo     = (const float*)d_in[14];
    const float* ln2_s  = (const float*)d_in[15];
    const float* ln2_b  = (const float*)d_in[16];
    const float* W1     = (const float*)d_in[17];
    const float* b1     = (const float*)d_in[18];
    const float* W2     = (const float*)d_in[19];
    const float* b2     = (const float*)d_in[20];
    float* out = (float*)d_out;

    char* ws = (char*)d_ws;
    size_t off = 0;
    auto take = [&](size_t bytes) -> char* {
        char* p = ws + off;
        off += (bytes + 255) & ~(size_t)255;
        return p;
    };
    unsigned short* WfxP = (unsigned short*)take((size_t)CC * CC * 2);
    unsigned short* WxP  = (unsigned short*)take((size_t)CC * CC * 2);
    unsigned short* WslP = (unsigned short*)take((size_t)DDIM * GG * 2);
    unsigned short* WoP  = (unsigned short*)take((size_t)CC * CC * 2);
    unsigned short* W1P  = (unsigned short*)take((size_t)CC * FFH * 2);
    unsigned short* W2P  = (unsigned short*)take((size_t)FFH * CC * 2);
    unsigned short* osP  = (unsigned short*)take((size_t)BB * HHD * 2048 * 2);
    float* stnum = (float*)take((size_t)BB * HHD * GG * DDIM * 4);
    float* stden = (float*)take((size_t)BB * HHD * GG * 4);
    unsigned short* swG = (unsigned short*)take((size_t)BB * HHD * NN * GG * 2);
    if (off > ws_size) return;  // workspace too small: refuse to scribble OOB

    (void)hipMemsetAsync(stnum, 0, (size_t)BB * HHD * GG * DDIM * 4, stream);
    (void)hipMemsetAsync(stden, 0, (size_t)BB * HHD * GG * 4, stream);

    // weight repacks (bf16, B-fragment layout)
    k_repack<<<dim3((CC * CC + 255) / 256), 256, 0, stream>>>(Wfx, WfxP, CC, CC);
    k_repack<<<dim3((CC * CC + 255) / 256), 256, 0, stream>>>(Wx, WxP, CC, CC);
    k_repack<<<dim3((DDIM * GG + 255) / 256), 256, 0, stream>>>(Wslice, WslP, DDIM, GG);
    k_repack<<<dim3((CC * CC + 255) / 256), 256, 0, stream>>>(Wo, WoP, CC, CC);
    k_repack<<<dim3((CC * FFH + 255) / 256), 256, 0, stream>>>(W1, W1P, CC, FFH);
    k_repack<<<dim3((FFH * CC + 255) / 256), 256, 0, stream>>>(W2, W2P, FFH, CC);

    const dim3 gridTok(NN / 32, BB);
    const size_t lds1 = (size_t)(32 * 264 + 32 * 264 + 8 * 32 * 64) * 2 + (size_t)32 * 64 * 4;
    const size_t lds2 = (size_t)(2048 * 5 + 4096) * 4;
    const size_t lds3 = (size_t)32 * 264 * 2;
    const size_t lds4 = (size_t)(32 * 264 + 32 * 1032) * 2;

    k_attn_pre<<<gridTok, 256, lds1, stream>>>(x, ln1_s, ln1_b, WxP, bx, WfxP, bfx,
                                               WslP, bslice, temp, swG, stnum, stden);
    k_attn_mid<<<dim3(BB * HHD), 64, lds2, stream>>>(stnum, stden, Wq, Wk, Wv, osP);
    k_attn_post<<<gridTok, 256, lds3, stream>>>(swG, osP, WoP, bo, x, out);
    k_ffn<<<gridTok, 256, lds4, stream>>>(ln2_s, ln2_b, W1P, b1, W2P, b2, out);

    (void)in_sizes; (void)n_in; (void)out_size;
}
